// HFOpenMoeSparseMLP_64089501991565
// MI455X (gfx1250) — compile-verified, matrix operand order checked by reading, not run on hardware
//
#include <hip/hip_runtime.h>
#include <hip/hip_bf16.h>

#define S_TOK 4096
#define HDIM  1024
#define FDIM  4096
#define NEXP  8
#define CAP   1280   // floor(2*1.25*4096/8), even, >= 4

typedef __attribute__((ext_vector_type(16))) __bf16 v16bf;
typedef __attribute__((ext_vector_type(8)))  float  v8f;

union FragU { unsigned u[8]; v16bf v; };

__device__ __forceinline__ unsigned short f2bf(float x) {
    unsigned u = __float_as_uint(x);
    unsigned r = u + 0x7FFFu + ((u >> 16) & 1u);   // round-to-nearest-even
    return (unsigned short)(r >> 16);
}

__device__ __forceinline__ v8f v8f_zero() {
    v8f z;
#pragma unroll
    for (int i = 0; i < 8; ++i) z[i] = 0.0f;
    return z;
}

// Load one 16x32 bf16 WMMA operand fragment from an LDS tile stored as
// [row][k] (pitch in bf16 elements, 32 K values per row).
// Per ISA 7.12.2: lanes 0-15 hold row=row0+lane, K {0..7, 16..23};
// lanes 16-31 hold row=row0+(lane-16), K {8..15, 24..31}.
__device__ __forceinline__ v16bf frag_ld(const unsigned short* base, int row0, int pitch) {
    const int lane = threadIdx.x & 31;
    const unsigned short* p = base + (row0 + (lane & 15)) * pitch + ((lane >> 4) << 3);
    FragU f;
    const unsigned* lo = (const unsigned*)p;          // K kb..kb+7   (16B)
    const unsigned* hi = (const unsigned*)(p + 16);   // K kb+16..kb+23
#pragma unroll
    for (int i = 0; i < 4; ++i) { f.u[i] = lo[i]; f.u[4 + i] = hi[i]; }
    return f.v;
}

__device__ __forceinline__ v8f wmma_bf16(v16bf a, v16bf b, v8f c) {
    return __builtin_amdgcn_wmma_f32_16x16x32_bf16(false, a, false, b, (short)0, c, false, false);
}

// ---------------------------------------------------------------------------
// 1. Router: logits -> softmax -> top1/top2 probs. One wave per token.
// ---------------------------------------------------------------------------
__global__ __launch_bounds__(256)
void k_router(const float* __restrict__ x, const float* __restrict__ gw,
              int* __restrict__ top1, int* __restrict__ top2,
              float* __restrict__ w1, float* __restrict__ w2) {
    const int wave = threadIdx.x >> 5;
    const int lane = threadIdx.x & 31;
    const int s = blockIdx.x * 8 + wave;
    const float* xr = x + (size_t)s * HDIM;

    float acc[NEXP];
#pragma unroll
    for (int e = 0; e < NEXP; ++e) acc[e] = 0.0f;

    for (int h = lane; h < HDIM; h += 32) {
        const float xv = xr[h];
        const float* g = gw + h * NEXP;
#pragma unroll
        for (int e = 0; e < NEXP; ++e) acc[e] += xv * g[e];
    }
#pragma unroll
    for (int off = 16; off; off >>= 1)
#pragma unroll
        for (int e = 0; e < NEXP; ++e) acc[e] += __shfl_xor(acc[e], off, 32);

    if (lane == 0) {
        float mx = acc[0];
#pragma unroll
        for (int e = 1; e < NEXP; ++e) mx = fmaxf(mx, acc[e]);
        float p[NEXP], se = 0.0f;
#pragma unroll
        for (int e = 0; e < NEXP; ++e) { p[e] = __expf(acc[e] - mx); se += p[e]; }
        const float inv = 1.0f / se;
        int b1 = 0; float v1 = -1.0f;
#pragma unroll
        for (int e = 0; e < NEXP; ++e) { p[e] *= inv; if (p[e] > v1) { v1 = p[e]; b1 = e; } }
        int b2 = 0; float v2 = -1.0f;
#pragma unroll
        for (int e = 0; e < NEXP; ++e) { if (e != b1 && p[e] > v2) { v2 = p[e]; b2 = e; } }
        top1[s] = b1; top2[s] = b2; w1[s] = v1; w2[s] = v2;
    }
}

// ---------------------------------------------------------------------------
// 2. Capacity scan: exact token-order cumsum per expert. Single block.
// ---------------------------------------------------------------------------
__global__ __launch_bounds__(256)
void k_scan(const int* __restrict__ top1, const int* __restrict__ top2,
            const float* __restrict__ w1, const float* __restrict__ w2,
            int* __restrict__ slot1, int* __restrict__ slot2,
            float* __restrict__ cw1, float* __restrict__ cw2,
            int* __restrict__ tok_of_slot) {
    __shared__ int c1[256][NEXP];
    __shared__ int c2[256][NEXP];
    __shared__ int tot1[NEXP];
    const int t = threadIdx.x;

    for (int i = t; i < NEXP * CAP; i += 256) tok_of_slot[i] = -1;

    int l1[NEXP], l2[NEXP];
#pragma unroll
    for (int e = 0; e < NEXP; ++e) { l1[e] = 0; l2[e] = 0; }
    const int base = t * 16;
    for (int i = 0; i < 16; ++i) { l1[top1[base + i]]++; l2[top2[base + i]]++; }
#pragma unroll
    for (int e = 0; e < NEXP; ++e) { c1[t][e] = l1[e]; c2[t][e] = l2[e]; }
    __syncthreads();

    if (t < NEXP) {
        int run = 0;
        for (int j = 0; j < 256; ++j) { int tmp = c1[j][t]; c1[j][t] = run; run += tmp; }
        tot1[t] = run;
        run = 0;
        for (int j = 0; j < 256; ++j) { int tmp = c2[j][t]; c2[j][t] = run; run += tmp; }
    }
    __syncthreads();

#pragma unroll
    for (int e = 0; e < NEXP; ++e) { l1[e] = 0; l2[e] = 0; }
    for (int i = 0; i < 16; ++i) {
        const int s = base + i;
        const int e1 = top1[s];
        const int r1 = c1[t][e1] + l1[e1]++;
        if (r1 < CAP) { slot1[s] = r1; cw1[s] = w1[s]; tok_of_slot[e1 * CAP + r1] = s; }
        else          { slot1[s] = 0;  cw1[s] = 0.0f; }
        const int e2 = top2[s];
        const int r2 = tot1[e2] + c2[t][e2] + l2[e2]++;
        if (r2 < CAP) { slot2[s] = r2; cw2[s] = w2[s]; tok_of_slot[e2 * CAP + r2] = s; }
        else          { slot2[s] = 0;  cw2[s] = 0.0f; }
    }
}

// ---------------------------------------------------------------------------
// 3. Dispatch gather: [E,C,H] bf16 rows (zeros for unfilled slots).
// ---------------------------------------------------------------------------
__global__ __launch_bounds__(256)
void k_dispatch(const float* __restrict__ x, const int* __restrict__ tok_of_slot,
                unsigned short* __restrict__ disp) {
    const int slot = blockIdx.x;
    const int tok = tok_of_slot[slot];
    unsigned short* dst = disp + (size_t)slot * HDIM;
    if (tok >= 0) {
        const float* src = x + (size_t)tok * HDIM;
        for (int i = threadIdx.x; i < HDIM; i += blockDim.x) dst[i] = f2bf(src[i]);
    } else {
        for (int i = threadIdx.x; i < HDIM; i += blockDim.x) dst[i] = 0;
    }
}

// ---------------------------------------------------------------------------
// 4. Fused gate+up+SwiGLU: ACT[e,c,f] = x1 * (x2*sigmoid(x2)) * u, bf16 out.
//    Block tile 128x64, 8 waves as 4(M) x 2(N); each wave owns a 32x32 tile
//    per output matrix: 2 A-frags x 2 B-frags -> 12 WMMAs per K-step per wave
//    against 16 ds_load_b128 (1.33 loads/WMMA). bf16 WMMA, f32 accumulate.
// ---------------------------------------------------------------------------
__global__ __launch_bounds__(256)
void k_gateup(const unsigned short* __restrict__ disp,
              const float* __restrict__ wg,
              const float* __restrict__ wu,
              unsigned short* __restrict__ act) {
    __shared__ unsigned short sA [128 * 40];
    __shared__ unsigned short sG1[64 * 40];
    __shared__ unsigned short sG2[64 * 40];
    __shared__ unsigned short sU [64 * 40];

    const int e  = blockIdx.z;
    const int c0 = blockIdx.y * 128;
    const int n0 = blockIdx.x * 64;

    const unsigned short* A = disp + (size_t)e * CAP * HDIM;
    const float* Wg = wg + (size_t)e * HDIM * (2 * FDIM);
    const float* Wu = wu + (size_t)e * HDIM * FDIM;

    const int w  = threadIdx.x >> 5;
    const int wm = (w & 3) * 32;     // 4 M-waves * 32 rows
    const int wn = (w >> 2) * 32;    // 2 N-waves * 32 cols

    v8f acc1[2][2], acc2[2][2], accu[2][2];   // [mi][nj]
#pragma unroll
    for (int mi = 0; mi < 2; ++mi)
#pragma unroll
        for (int nj = 0; nj < 2; ++nj) {
            acc1[mi][nj] = v8f_zero(); acc2[mi][nj] = v8f_zero(); accu[mi][nj] = v8f_zero();
        }

    for (int k0 = 0; k0 < HDIM; k0 += 32) {
        {   // A tile: 128 rows x 32 K bf16; 2 threads per row, 32B each
            const int row = threadIdx.x >> 1;
            const int kq  = (threadIdx.x & 1) << 4;
            const unsigned* src = (const unsigned*)(A + (size_t)(c0 + row) * HDIM + k0 + kq);
            unsigned* dst = (unsigned*)(&sA[row * 40 + kq]);
#pragma unroll
            for (int i = 0; i < 8; ++i) dst[i] = src[i];
        }
        {   // B tiles: fp32 -> bf16, transposed to [n][k]
            const int k  = threadIdx.x >> 3;
            const int n8 = (threadIdx.x & 7) << 3;
            const float* s1 = Wg + (size_t)(k0 + k) * (2 * FDIM) + n0 + n8;
            const float* s2 = s1 + FDIM;
            const float* s3 = Wu + (size_t)(k0 + k) * FDIM + n0 + n8;
            if (k0 + 32 < HDIM) {   // pull next K-step weight rows toward L2/L0
                __builtin_prefetch(s1 + (size_t)32 * (2 * FDIM), 0, 0);
                __builtin_prefetch(s2 + (size_t)32 * (2 * FDIM), 0, 0);
                __builtin_prefetch(s3 + (size_t)32 * FDIM, 0, 0);
            }
#pragma unroll
            for (int j = 0; j < 8; ++j) sG1[(n8 + j) * 40 + k] = f2bf(s1[j]);
#pragma unroll
            for (int j = 0; j < 8; ++j) sG2[(n8 + j) * 40 + k] = f2bf(s2[j]);
#pragma unroll
            for (int j = 0; j < 8; ++j) sU [(n8 + j) * 40 + k] = f2bf(s3[j]);
        }
        __syncthreads();

        v16bf af[2];
        af[0] = frag_ld(sA, wm, 40);
        af[1] = frag_ld(sA, wm + 16, 40);
#pragma unroll
        for (int nj = 0; nj < 2; ++nj) {
            v16bf b1 = frag_ld(sG1, wn + nj * 16, 40);
            acc1[0][nj] = wmma_bf16(af[0], b1, acc1[0][nj]);
            acc1[1][nj] = wmma_bf16(af[1], b1, acc1[1][nj]);
            v16bf b2 = frag_ld(sG2, wn + nj * 16, 40);
            acc2[0][nj] = wmma_bf16(af[0], b2, acc2[0][nj]);
            acc2[1][nj] = wmma_bf16(af[1], b2, acc2[1][nj]);
            v16bf bu = frag_ld(sU, wn + nj * 16, 40);
            accu[0][nj] = wmma_bf16(af[0], bu, accu[0][nj]);
            accu[1][nj] = wmma_bf16(af[1], bu, accu[1][nj]);
        }
        __syncthreads();
    }

    // Epilogue: SwiGLU, write bf16. C/D layout: VGPR i -> M=i (lanes 0-15) or
    // M=i+8 (lanes 16-31), N=lane&15.
    const int lane  = threadIdx.x & 31;
    const int mhalf = (lane >> 4) << 3;
    const int ncol  = lane & 15;
#pragma unroll
    for (int mi = 0; mi < 2; ++mi) {
#pragma unroll
        for (int nj = 0; nj < 2; ++nj) {
#pragma unroll
            for (int i = 0; i < 8; ++i) {
                const float x1 = acc1[mi][nj][i];
                const float x2 = acc2[mi][nj][i];
                const float uu = accu[mi][nj][i];
                const float sig = __builtin_amdgcn_rcpf(1.0f + __expf(-x2));
                const float v = x1 * (x2 * sig) * uu;
                const int m = c0 + wm + mi * 16 + mhalf + i;
                const int n = n0 + wn + nj * 16 + ncol;
                act[((size_t)e * CAP + m) * FDIM + n] = f2bf(v);
            }
        }
    }
}

// ---------------------------------------------------------------------------
// 5. Down projection: EO[e,c,h] = ACT[e] (C x F, bf16) @ Wd[e] (F x H), f32 out.
//    Same 128x64 block tile, 32x32 per wave (4 WMMAs / 8 b128 per K-step).
// ---------------------------------------------------------------------------
__global__ __launch_bounds__(256)
void k_down(const unsigned short* __restrict__ act,
            const float* __restrict__ wd,
            float* __restrict__ eo) {
    __shared__ unsigned short sA[128 * 40];
    __shared__ unsigned short sB[64 * 40];

    const int e  = blockIdx.z;
    const int c0 = blockIdx.y * 128;
    const int n0 = blockIdx.x * 64;

    const unsigned short* A = act + (size_t)e * CAP * FDIM;
    const float* B = wd + (size_t)e * FDIM * HDIM;

    const int w  = threadIdx.x >> 5;
    const int wm = (w & 3) * 32;
    const int wn = (w >> 2) * 32;

    v8f acc[2][2];
#pragma unroll
    for (int mi = 0; mi < 2; ++mi)
#pragma unroll
        for (int nj = 0; nj < 2; ++nj) acc[mi][nj] = v8f_zero();

    for (int k0 = 0; k0 < FDIM; k0 += 32) {
        {
            const int row = threadIdx.x >> 1;
            const int kq  = (threadIdx.x & 1) << 4;
            const unsigned* src = (const unsigned*)(A + (size_t)(c0 + row) * FDIM + k0 + kq);
            unsigned* dst = (unsigned*)(&sA[row * 40 + kq]);
#pragma unroll
            for (int i = 0; i < 8; ++i) dst[i] = src[i];
        }
        {
            const int k  = threadIdx.x >> 3;
            const int n8 = (threadIdx.x & 7) << 3;
            const float* s = B + (size_t)(k0 + k) * HDIM + n0 + n8;
            if (k0 + 32 < FDIM)
                __builtin_prefetch(s + (size_t)32 * HDIM, 0, 0);
#pragma unroll
            for (int j = 0; j < 8; ++j) sB[(n8 + j) * 40 + k] = f2bf(s[j]);
        }
        __syncthreads();

        v16bf af[2];
        af[0] = frag_ld(sA, wm, 40);
        af[1] = frag_ld(sA, wm + 16, 40);
#pragma unroll
        for (int nj = 0; nj < 2; ++nj) {
            v16bf b = frag_ld(sB, wn + nj * 16, 40);
            acc[0][nj] = wmma_bf16(af[0], b, acc[0][nj]);
            acc[1][nj] = wmma_bf16(af[1], b, acc[1][nj]);
        }
        __syncthreads();
    }

    const int lane  = threadIdx.x & 31;
    const int mhalf = (lane >> 4) << 3;
    const int ncol  = lane & 15;
#pragma unroll
    for (int mi = 0; mi < 2; ++mi)
#pragma unroll
        for (int nj = 0; nj < 2; ++nj)
#pragma unroll
            for (int i = 0; i < 8; ++i) {
                const int m = c0 + wm + mi * 16 + mhalf + i;
                const int n = n0 + wn + nj * 16 + ncol;
                eo[((size_t)e * CAP + m) * HDIM + n] = acc[mi][nj][i];
            }
}

// ---------------------------------------------------------------------------
// 6. Combine: out[s] = cw1*EO[e1,slot1] + cw2*EO[e2,slot2]
// ---------------------------------------------------------------------------
__global__ __launch_bounds__(256)
void k_combine(const float* __restrict__ eo,
               const int* __restrict__ top1, const int* __restrict__ top2,
               const int* __restrict__ slot1, const int* __restrict__ slot2,
               const float* __restrict__ cw1, const float* __restrict__ cw2,
               float* __restrict__ out) {
    const int s = blockIdx.x;
    const float a = cw1[s];
    const float b = cw2[s];
    const float* r1 = eo + ((size_t)(top1[s] * CAP + slot1[s])) * HDIM;
    const float* r2 = eo + ((size_t)(top2[s] * CAP + slot2[s])) * HDIM;
    float* o = out + (size_t)s * HDIM;
    for (int i = threadIdx.x; i < HDIM; i += blockDim.x)
        o[i] = a * r1[i] + b * r2[i];
}

extern "C" void kernel_launch(void* const* d_in, const int* in_sizes, int n_in,
                              void* d_out, int out_size, void* d_ws, size_t ws_size,
                              hipStream_t stream) {
    (void)in_sizes; (void)n_in; (void)out_size; (void)ws_size;

    const float* hs = (const float*)d_in[0];   // [S,H]
    const float* gw = (const float*)d_in[1];   // [H,E]
    const float* wg = (const float*)d_in[2];   // [E,H,2F]
    const float* wu = (const float*)d_in[3];   // [E,H,F]
    const float* wd = (const float*)d_in[4];   // [E,F,H]
    float* out = (float*)d_out;                // [S,H]

    char* ws = (char*)d_ws;
    size_t off = 0;
    auto take = [&](size_t bytes) -> char* {
        char* p = ws + off;
        off += (bytes + 255) & ~(size_t)255;
        return p;
    };
    int*   top1  = (int*)take((size_t)S_TOK * 4);
    int*   top2  = (int*)take((size_t)S_TOK * 4);
    float* w1    = (float*)take((size_t)S_TOK * 4);
    float* w2    = (float*)take((size_t)S_TOK * 4);
    int*   slot1 = (int*)take((size_t)S_TOK * 4);
    int*   slot2 = (int*)take((size_t)S_TOK * 4);
    float* cw1   = (float*)take((size_t)S_TOK * 4);
    float* cw2   = (float*)take((size_t)S_TOK * 4);
    int*   tos   = (int*)take((size_t)NEXP * CAP * 4);
    unsigned short* disp = (unsigned short*)take((size_t)NEXP * CAP * HDIM * 2);
    unsigned short* act  = (unsigned short*)take((size_t)NEXP * CAP * FDIM * 2);
    float* eo    = (float*)take((size_t)NEXP * CAP * HDIM * 4);

    k_router<<<S_TOK / 8, 256, 0, stream>>>(hs, gw, top1, top2, w1, w2);
    k_scan<<<1, 256, 0, stream>>>(top1, top2, w1, w2, slot1, slot2, cw1, cw2, tos);
    k_dispatch<<<NEXP * CAP, 256, 0, stream>>>(hs, tos, disp);
    k_gateup<<<dim3(FDIM / 64, CAP / 128, NEXP), 256, 0, stream>>>(disp, wg, wu, act);
    k_down<<<dim3(HDIM / 64, CAP / 128, NEXP), 256, 0, stream>>>(act, wd, eo);
    k_combine<<<S_TOK, 256, 0, stream>>>(eo, top1, top2, slot1, slot2, cw1, cw2, out);
}
